// NaivePolynomialKANLayer_37142877176051
// MI455X (gfx1250) — matrix operand order repaired and checked by reading, not compiled, not run on hardware
//
#include <hip/hip_runtime.h>
#include <hip/hip_bf16.h>

typedef __attribute__((ext_vector_type(16))) __bf16 v16bf;
typedef __attribute__((ext_vector_type(8)))  float  v8f;

#define N_NODES   50000
#define IN_FEATS  128
#define OUT_FEATS 128
#define DEG_P1    4
#define KDIM      (IN_FEATS * DEG_P1)       // 512
#define M_TILES   (N_NODES / 16)            // 3125 (exact)
#define PACKB_ELEMS (KDIM * OUT_FEATS)      // 65536 bf16
#define PACKB_BYTES (PACKB_ELEMS * 2)       // 131072 B

// ---------------------------------------------------------------------------
// Kernel 0: repack coeffs[O][I][4] (f32) into bf16 WMMA B-fragment layout.
// B is K x N = 512 x 128 where K = i*4 + d, N = o; B[k][n] = coeffs[n*512 + k].
// Fragment layout (ISA 16-bit B 32x16, wave32): per (kt, nt) tile, lane l
// holds 16 bf16: N = nt*16 + (l&15), K = kt*32 + (l>>4)*8 + (e<8 ? e : e+8), e = 0..15.
// Stored contiguously: off = ((kt*8 + nt)*32 + lane)*16 + e
// ---------------------------------------------------------------------------
__global__ void pack_coeffs(const float* __restrict__ coeffs,
                            __bf16* __restrict__ packed) {
    int off = blockIdx.x * blockDim.x + threadIdx.x;
    if (off >= PACKB_ELEMS) return;
    int e    = off & 15;
    int lane = (off >> 4) & 31;
    int nt   = (off >> 9) & 7;
    int kt   = off >> 12;
    int K = kt * 32 + (lane >> 4) * 8 + (e < 8 ? e : e + 8);
    int N = nt * 16 + (lane & 15);
    packed[off] = (__bf16)coeffs[N * KDIM + K];
}

// ---------------------------------------------------------------------------
// Kernel 1: t[n][o] = sum_k P[n][k] * B[k][o] using v_wmma_f32_16x16x32_bf16.
// 8 waves/block, one 16-row M-tile per wave, all 8 N-tiles accumulated in
// registers. Packed B (128 KB) staged in LDS once per block.
// A fragment (ISA 16-bit A 16x32): lane = M (l&15); lane group g = l>>4:
//   element e -> K = kt*32 + g*8 + (e<8 ? e : e+8)
// => per lane needs features i0, i0+1, i0+4, i0+5 with i0 = kt*8 + g*2,
//    each contributing powers x^0..x^3.
// ---------------------------------------------------------------------------
__global__ void __launch_bounds__(256) node_transform(
    const float* __restrict__ x,
    const __bf16* __restrict__ packedB,
    float* __restrict__ t) {
    __shared__ __bf16 ldsB[PACKB_ELEMS];   // 128 KB (<= 320 KB WGP LDS)

    // Cooperative global -> LDS copy (b128 per lane)
    {
        const uint4* g = (const uint4*)packedB;
        uint4* s = (uint4*)ldsB;
        for (int i = threadIdx.x; i < PACKB_BYTES / 16; i += 256) s[i] = g[i];
    }
    __syncthreads();

    const int wave = threadIdx.x >> 5;
    const int lane = threadIdx.x & 31;
    const int mtile = blockIdx.x * 8 + wave;
    if (mtile >= M_TILES) return;          // wave-uniform: EXEC stays all-ones

    const int g16 = lane >> 4;             // 0 or 1
    const int m   = lane & 15;
    const float* xrow = x + (size_t)(mtile * 16 + m) * IN_FEATS;

    v8f acc[8];
#pragma unroll
    for (int nt = 0; nt < 8; ++nt) acc[nt] = (v8f){0, 0, 0, 0, 0, 0, 0, 0};

    const uint4* ldsq = (const uint4*)ldsB;

    for (int kt = 0; kt < 16; ++kt) {
        const int i0 = kt * 8 + g16 * 2;
        float2 xa = *(const float2*)(xrow + i0);       // features i0, i0+1
        float2 xb = *(const float2*)(xrow + i0 + 4);   // features i0+4, i0+5
        float v0 = xa.x, v1 = xa.y, v4 = xb.x, v5 = xb.y;

        v16bf a;
        a[0]  = (__bf16)1.0f; a[1]  = (__bf16)v0; a[2]  = (__bf16)(v0 * v0); a[3]  = (__bf16)(v0 * v0 * v0);
        a[4]  = (__bf16)1.0f; a[5]  = (__bf16)v1; a[6]  = (__bf16)(v1 * v1); a[7]  = (__bf16)(v1 * v1 * v1);
        a[8]  = (__bf16)1.0f; a[9]  = (__bf16)v4; a[10] = (__bf16)(v4 * v4); a[11] = (__bf16)(v4 * v4 * v4);
        a[12] = (__bf16)1.0f; a[13] = (__bf16)v5; a[14] = (__bf16)(v5 * v5); a[15] = (__bf16)(v5 * v5 * v5);

#pragma unroll
        for (int nt = 0; nt < 8; ++nt) {
            int qi = ((kt * 8 + nt) * 32 + lane) * 2;  // uint4 index into LDS
            union { uint4 q[2]; v16bf v; } bb;
            bb.q[0] = ldsq[qi];
            bb.q[1] = ldsq[qi + 1];
            acc[nt] = __builtin_amdgcn_wmma_f32_16x16x32_bf16(
                false, a, false, bb.v, (short)0, acc[nt], false, false);
        }
    }

    // C/D layout: lanes 0-15 -> M=r, N=lane; lanes 16-31 -> M=8+r, N=lane-16
    const int rowbase = mtile * 16 + g16 * 8;
#pragma unroll
    for (int nt = 0; nt < 8; ++nt) {
        int col = nt * 16 + m;
#pragma unroll
        for (int r = 0; r < 8; ++r)
            t[(size_t)(rowbase + r) * OUT_FEATS + col] = acc[nt][r];
    }
}

// ---------------------------------------------------------------------------
// Kernel 2: h[n][o] = bias[o]
// ---------------------------------------------------------------------------
__global__ void init_bias(const float* __restrict__ bias,
                          float* __restrict__ h, int total) {
    int i = blockIdx.x * blockDim.x + threadIdx.x;
    if (i < total) h[i] = bias[i & (OUT_FEATS - 1)];
}

// ---------------------------------------------------------------------------
// Kernel 3: h[dst[e]] += t[src[e]] — one wave per edge, float4 per lane,
// native global_atomic_add_f32. t and h both fit in the 192 MB L2.
// ---------------------------------------------------------------------------
__global__ void edge_scatter(const float* __restrict__ t,
                             const int* __restrict__ src,
                             const int* __restrict__ dst,
                             float* __restrict__ h, int E) {
    int wave = (int)((blockIdx.x * blockDim.x + threadIdx.x) >> 5);
    int lane = threadIdx.x & 31;
    if (wave >= E) return;                 // wave-uniform
    int s = src[wave];
    int d = dst[wave];
    const float4* ts = (const float4*)(t + (size_t)s * OUT_FEATS);
    float4 v = ts[lane];
    float* hp = h + (size_t)d * OUT_FEATS + lane * 4;
    unsafeAtomicAdd(hp + 0, v.x);
    unsafeAtomicAdd(hp + 1, v.y);
    unsafeAtomicAdd(hp + 2, v.z);
    unsafeAtomicAdd(hp + 3, v.w);
}

// ---------------------------------------------------------------------------
extern "C" void kernel_launch(void* const* d_in, const int* in_sizes, int n_in,
                              void* d_out, int out_size, void* d_ws, size_t ws_size,
                              hipStream_t stream) {
    const float* x      = (const float*)d_in[0];
    const float* coeffs = (const float*)d_in[1];
    const float* bias   = (const float*)d_in[2];
    const int*   src    = (const int*)d_in[3];
    const int*   dst    = (const int*)d_in[4];
    float* h = (float*)d_out;
    const int E = in_sizes[3];

    __bf16* packedB = (__bf16*)d_ws;
    float*  t       = (float*)((char*)d_ws + PACKB_BYTES);

    pack_coeffs<<<(PACKB_ELEMS + 255) / 256, 256, 0, stream>>>(coeffs, packedB);
    node_transform<<<(M_TILES + 7) / 8, 256, 0, stream>>>(x, packedB, t);
    init_bias<<<(N_NODES * OUT_FEATS + 255) / 256, 256, 0, stream>>>(
        bias, h, N_NODES * OUT_FEATS);
    edge_scatter<<<(unsigned)(((size_t)E * 32 + 255) / 256), 256, 0, stream>>>(
        t, src, dst, h, E);
}
